// TransformerBlock_72739566125760
// MI455X (gfx1250) — matrix operand order repaired and checked by reading, not compiled, not run on hardware
//
#include <hip/hip_runtime.h>
#include <hip/hip_bf16.h>

// ---------------------------------------------------------------------------
// Shapes (compile-time, match the reference)
// ---------------------------------------------------------------------------
#define SEQ   2048
#define DIM_  2560
#define NHEAD 32
#define HD    80       // head dim
#define HDP   96       // head dim padded to multiple of 32 for WMMA K
#define ROT   32
#define INTER_ 10240

typedef __attribute__((ext_vector_type(16))) __bf16 v16bf;
typedef __attribute__((ext_vector_type(8)))  __bf16 v8bf;
typedef __attribute__((ext_vector_type(4)))  __bf16 v4bf;
typedef __attribute__((ext_vector_type(8)))  float  v8f;

__device__ __forceinline__ v8f vzero8() {
    v8f z = {0.f,0.f,0.f,0.f,0.f,0.f,0.f,0.f};
    return z;
}

__device__ __forceinline__ v16bf mk_frag(v8bf lo, v8bf hi) {
    return __builtin_shufflevector(lo, hi, 0,1,2,3,4,5,6,7,8,9,10,11,12,13,14,15);
}

__device__ __forceinline__ v8f wmma_bf16(v16bf a, v16bf b, v8f c) {
    // D(f32 16x16) = A(bf16 16x32) * B(bf16 32x16) + C
    return __builtin_amdgcn_wmma_f32_16x16x32_bf16(
        /*neg_a=*/false, a, /*neg_b=*/false, b,
        /*c_mod=*/(short)0, c, /*reuse_a=*/false, /*reuse_b=*/false);
}

__device__ __forceinline__ float gelu_tanh(float x) {
    float x3 = x * x * x;
    return 0.5f * x * (1.f + tanhf(0.7978845608028654f * (x + 0.044715f * x3)));
}

__device__ __forceinline__ float redmax16(float v) {
    #pragma unroll
    for (int m = 1; m < 16; m <<= 1) v = fmaxf(v, __shfl_xor(v, m, 32));
    return v;
}
__device__ __forceinline__ float redsum16(float v) {
    #pragma unroll
    for (int m = 1; m < 16; m <<= 1) v += __shfl_xor(v, m, 32);
    return v;
}

// Async global -> LDS copy of 16 bytes (CDNA5, tracked by ASYNCcnt).
// LDS destination = low 32 bits of the generic shared-space address
// (ISA 10.2).
__device__ __forceinline__ void async_copy_b128(const void* gsrc, void* lds_dst) {
    const unsigned ldsoff = (unsigned)(uintptr_t)lds_dst;
    asm volatile("global_load_async_to_lds_b128 %0, %1, off"
                 :: "v"(ldsoff), "v"(gsrc) : "memory");
}
__device__ __forceinline__ void wait_asynccnt0() {
    asm volatile("s_wait_asynccnt 0" ::: "memory");
}

// ---------------------------------------------------------------------------
// LayerNorm: x[S,D] f32 -> h[S,D] bf16
// ---------------------------------------------------------------------------
__global__ __launch_bounds__(256) void ln_kernel(const float* __restrict__ x,
                                                 const float* __restrict__ w,
                                                 const float* __restrict__ b,
                                                 __bf16* __restrict__ h) {
    __shared__ float s1[256], s2[256];
    const int row = blockIdx.x;
    const int tid = threadIdx.x;
    const float* xr = x + (size_t)row * DIM_;
    float a = 0.f, q = 0.f;
    for (int i = tid; i < DIM_; i += 256) {
        float v = xr[i];
        a += v; q += v * v;
    }
    s1[tid] = a; s2[tid] = q;
    __syncthreads();
    #pragma unroll
    for (int s = 128; s > 0; s >>= 1) {
        if (tid < s) { s1[tid] += s1[tid + s]; s2[tid] += s2[tid + s]; }
        __syncthreads();
    }
    const float mu  = s1[0] * (1.f / DIM_);
    const float var = s2[0] * (1.f / DIM_) - mu * mu;
    const float inv = rsqrtf(var + 1e-5f);
    __bf16* hr = h + (size_t)row * DIM_;
    for (int i = tid; i < DIM_; i += 256) {
        hr[i] = (__bf16)((xr[i] - mu) * inv * w[i] + b[i]);
    }
}

// ---------------------------------------------------------------------------
// Generic WMMA GEMM: C[M,N] = A[M,K](bf16) * W[N,K]^T(f32->bf16) + bias
// MODE 0: store f32          MODE 1: gelu -> bf16
// MODE 2: store f32 (+res1 +res2)   (final residual fusion)
// BM=BN=128, BK=32, 256 threads = 8 waves; wave w computes rows 16w..16w+15.
// Double-buffered LDS; A tile staged via GLOBAL_LOAD_ASYNC_TO_LDS_B128
// (ASYNCcnt), W tile converted f32->bf16 through VGPRs.
// ---------------------------------------------------------------------------
#define BK 32
#define LDSS 40   // padded row stride (bf16 elements)

template <int MODE>
__global__ __launch_bounds__(256) void gemm_kernel(
    const __bf16* __restrict__ A, const float* __restrict__ W,
    const float* __restrict__ bias, void* __restrict__ out,
    const float* __restrict__ res1, const float* __restrict__ res2,
    int M, int N, int K)
{
    __shared__ alignas(16) __bf16 As[2][128 * LDSS];
    __shared__ alignas(16) __bf16 Bs[2][128 * LDSS];

    const int tid  = threadIdx.x;
    const int bm   = blockIdx.y * 128;
    const int bn   = blockIdx.x * 128;
    const int wave = tid >> 5;
    const int lane = tid & 31;
    const int l15  = lane & 15;
    const int hi   = lane >> 4;
    const int koffA = hi << 3;   // A-operand K offset pattern
    const int koffB = hi << 3;   // B-operand K offset pattern (same as A)

    // per-thread staging coordinates
    const int ar0 = (tid * 2) >> 2;            // A chunk 0 row
    const int ac0 = ((tid * 2) & 3) << 3;      // A chunk 0 col
    const int ar1 = (tid * 2 + 1) >> 2;
    const int ac1 = ((tid * 2 + 1) & 3) << 3;

    v8f acc[8];
    #pragma unroll
    for (int i = 0; i < 8; ++i) acc[i] = vzero8();

    const int nK = K / BK;

    // ---- prologue: stage tile 0 into buffer 0 ----
    async_copy_b128(&A[(size_t)(bm + ar0) * K + ac0], &As[0][ar0 * LDSS + ac0]);
    async_copy_b128(&A[(size_t)(bm + ar1) * K + ac1], &As[0][ar1 * LDSS + ac1]);
    {
        #pragma unroll
        for (int i = 0; i < 4; ++i) {
            const int c   = tid * 4 + i;
            const int row = c >> 3;
            const int col = (c & 7) << 2;
            const float4 f =
                *reinterpret_cast<const float4*>(&W[(size_t)(bn + row) * K + col]);
            v4bf p;
            p[0] = (__bf16)f.x; p[1] = (__bf16)f.y;
            p[2] = (__bf16)f.z; p[3] = (__bf16)f.w;
            *reinterpret_cast<v4bf*>(&Bs[0][row * LDSS + col]) = p;
        }
    }
    wait_asynccnt0();
    __syncthreads();

    int buf = 0;
    for (int kt = 0; kt < nK; ++kt) {
        const bool more = (kt + 1) < nK;
        float4 breg[4];
        if (more) {
            const int kk = (kt + 1) * BK;
            // issue next A tile async into the other buffer (no VGPR traffic)
            async_copy_b128(&A[(size_t)(bm + ar0) * K + kk + ac0],
                            &As[buf ^ 1][ar0 * LDSS + ac0]);
            async_copy_b128(&A[(size_t)(bm + ar1) * K + kk + ac1],
                            &As[buf ^ 1][ar1 * LDSS + ac1]);
            // pull next W tile into registers (overlaps with WMMA below)
            #pragma unroll
            for (int i = 0; i < 4; ++i) {
                const int c   = tid * 4 + i;
                const int row = c >> 3;
                const int col = (c & 7) << 2;
                breg[i] = *reinterpret_cast<const float4*>(
                    &W[(size_t)(bn + row) * K + kk + col]);
            }
        }

        // ---- compute from current buffer: 1 A frag, 8 WMMAs per wave ----
        const int arow = (wave << 4) + l15;
        const v16bf af = mk_frag(
            *reinterpret_cast<const v8bf*>(&As[buf][arow * LDSS + koffA]),
            *reinterpret_cast<const v8bf*>(&As[buf][arow * LDSS + koffA + 16]));
        #pragma unroll
        for (int nt = 0; nt < 8; ++nt) {
            const int brow = (nt << 4) + l15;
            const v16bf bfrag = mk_frag(
                *reinterpret_cast<const v8bf*>(&Bs[buf][brow * LDSS + koffB]),
                *reinterpret_cast<const v8bf*>(&Bs[buf][brow * LDSS + koffB + 16]));
            acc[nt] = wmma_bf16(af, bfrag, acc[nt]);
        }

        if (more) {
            // convert + store next W tile into the other buffer
            #pragma unroll
            for (int i = 0; i < 4; ++i) {
                const int c   = tid * 4 + i;
                const int row = c >> 3;
                const int col = (c & 7) << 2;
                v4bf p;
                p[0] = (__bf16)breg[i].x; p[1] = (__bf16)breg[i].y;
                p[2] = (__bf16)breg[i].z; p[3] = (__bf16)breg[i].w;
                *reinterpret_cast<v4bf*>(&Bs[buf ^ 1][row * LDSS + col]) = p;
            }
        }
        wait_asynccnt0();   // next A tile resident
        __syncthreads();    // all waves done reading buf / writing buf^1
        buf ^= 1;
    }

    // ---- epilogue ----
    const int hi8 = hi << 3;
    #pragma unroll
    for (int nt = 0; nt < 8; ++nt) {
        const int col = bn + (nt << 4) + l15;
        const float bv = bias[col];
        #pragma unroll
        for (int r = 0; r < 8; ++r) {
            const int row = bm + (wave << 4) + r + hi8;
            const size_t idx = (size_t)row * N + col;
            float cv = acc[nt][r] + bv;
            if (MODE == 0) {
                ((float*)out)[idx] = cv;
            } else if (MODE == 1) {
                ((__bf16*)out)[idx] = (__bf16)gelu_tanh(cv);
            } else {
                ((float*)out)[idx] = cv + res1[idx] + res2[idx];
            }
        }
    }
}

// ---------------------------------------------------------------------------
// RoPE + layout: qkv[S,3D] f32 -> Q,K bf16 [H][S][96] (zero-padded), V^T bf16 [H][80][S]
// ---------------------------------------------------------------------------
__global__ __launch_bounds__(128) void rope_kernel(const float* __restrict__ qkv,
                                                   const int* __restrict__ pos,
                                                   __bf16* __restrict__ Q,
                                                   __bf16* __restrict__ Kd,
                                                   __bf16* __restrict__ Vt) {
    const int s = blockIdx.x;
    const int h = blockIdx.y;
    const int d = threadIdx.x;
    if (d >= HDP) return;
    const size_t qoff = (size_t)h * SEQ * HDP + (size_t)s * HDP + d;
    if (d >= HD) {  // zero pad so the padded K-dim contributes nothing
        Q[qoff] = (__bf16)0.f;
        Kd[qoff] = (__bf16)0.f;
        return;
    }
    const float* base = qkv + (size_t)s * (3 * DIM_) + (size_t)h * HD;
    float qv = base[d];
    float kv = base[DIM_ + d];
    const float vv = base[2 * DIM_ + d];
    if (d < ROT) {
        const int i = d & 15;                       // freq index (half = 16)
        const float inv = __powf(10000.f, -(float)i * (1.f / 16.f));
        const float ang = (float)pos[s] * inv;
        const float c = __cosf(ang), sn = __sinf(ang);
        const float qr = (d < 16) ? -base[d + 16] : base[d - 16];
        const float kr = (d < 16) ? -base[DIM_ + d + 16] : base[DIM_ + d - 16];
        qv = qv * c + qr * sn;
        kv = kv * c + kr * sn;
    }
    Q[qoff] = (__bf16)qv;
    Kd[qoff] = (__bf16)kv;
    Vt[(size_t)h * HD * SEQ + (size_t)d * SEQ + s] = (__bf16)vv;
}

// ---------------------------------------------------------------------------
// Flash attention (causal), WMMA. Block = (128 q rows, 1 head), 8 waves,
// wave owns 16 rows. Key steps of 32 (= two 16x16 score tiles -> one K=32
// P*V step). P redistributed D-layout -> A-layout through per-wave LDS.
// ---------------------------------------------------------------------------
__global__ __launch_bounds__(256) void attn_kernel(const __bf16* __restrict__ Q,
                                                   const __bf16* __restrict__ Kd,
                                                   const __bf16* __restrict__ Vt,
                                                   __bf16* __restrict__ Y) {
    __shared__ alignas(16) __bf16 Pl[8 * 16 * 40];  // per-wave 16x32 (stride 40)

    const int h    = blockIdx.y;
    const int qb   = blockIdx.x * 128;
    const int wave = threadIdx.x >> 5;
    const int lane = threadIdx.x & 31;
    const int l15  = lane & 15;
    const int hi   = lane >> 4;
    const int koffA = hi << 3;
    const int koffB = hi << 3;
    const int m0   = qb + wave * 16;
    const float scale = 0.11180339887498949f;  // 1/sqrt(80)

    const __bf16* qh = Q  + (size_t)h * SEQ * HDP;
    const __bf16* kh = Kd + (size_t)h * SEQ * HDP;
    const __bf16* vh = Vt + (size_t)h * HD * SEQ;
    __bf16* pbuf = &Pl[wave * 16 * 40];

    // Q fragments (3 x K=32 chunks over padded head dim 96)
    v16bf qf[3];
    #pragma unroll
    for (int c = 0; c < 3; ++c) {
        const size_t base = (size_t)(m0 + l15) * HDP + c * 32 + koffA;
        qf[c] = mk_frag(*reinterpret_cast<const v8bf*>(&qh[base]),
                        *reinterpret_cast<const v8bf*>(&qh[base + 16]));
    }

    v8f acc[5];
    #pragma unroll
    for (int i = 0; i < 5; ++i) acc[i] = vzero8();
    float mrow[8], lrow[8];
    #pragma unroll
    for (int r = 0; r < 8; ++r) { mrow[r] = -1e30f; lrow[r] = 0.f; }

    for (int j = 0; j < m0 + 16; j += 32) {
        // ---- scores: two 16x16 tiles ----
        v8f st[2];
        #pragma unroll
        for (int t = 0; t < 2; ++t) {
            st[t] = vzero8();
            const int krow = j + t * 16 + l15;
            #pragma unroll
            for (int c = 0; c < 3; ++c) {
                const size_t kb = (size_t)krow * HDP + c * 32 + koffB;
                const v16bf kf = mk_frag(*reinterpret_cast<const v8bf*>(&kh[kb]),
                                         *reinterpret_cast<const v8bf*>(&kh[kb + 16]));
                st[t] = wmma_bf16(qf[c], kf, st[t]);
            }
        }
        // ---- scale + causal mask ----
        #pragma unroll
        for (int t = 0; t < 2; ++t) {
            const int n = j + t * 16 + l15;
            #pragma unroll
            for (int r = 0; r < 8; ++r) {
                const int m = m0 + r + (hi << 3);
                float v = st[t][r] * scale;
                st[t][r] = (n > m) ? -1e30f : v;
            }
        }
        // ---- online softmax update ----
        float alpha[8];
        #pragma unroll
        for (int r = 0; r < 8; ++r) {
            const float rmax = redmax16(fmaxf(st[0][r], st[1][r]));
            const float nm = fmaxf(mrow[r], rmax);
            alpha[r] = __expf(mrow[r] - nm);
            mrow[r] = nm;
        }
        #pragma unroll
        for (int t = 0; t < 2; ++t)
            #pragma unroll
            for (int r = 0; r < 8; ++r)
                st[t][r] = __expf(st[t][r] - mrow[r]);
        #pragma unroll
        for (int r = 0; r < 8; ++r) {
            const float rs = redsum16(st[0][r] + st[1][r]);
            lrow[r] = lrow[r] * alpha[r] + rs;
        }
        #pragma unroll
        for (int n5 = 0; n5 < 5; ++n5)
            #pragma unroll
            for (int r = 0; r < 8; ++r)
                acc[n5][r] *= alpha[r];

        // ---- P: D-layout -> LDS -> A-layout fragment ----
        #pragma unroll
        for (int t = 0; t < 2; ++t)
            #pragma unroll
            for (int r = 0; r < 8; ++r)
                pbuf[(r + (hi << 3)) * 40 + t * 16 + l15] = (__bf16)st[t][r];
        asm volatile("s_wait_dscnt 0" ::: "memory");  // per-wave LDS RAW fence
        const v16bf pf = mk_frag(
            *reinterpret_cast<const v8bf*>(&pbuf[l15 * 40 + koffA]),
            *reinterpret_cast<const v8bf*>(&pbuf[l15 * 40 + koffA + 16]));

        // ---- acc += P * V  (5 N-tiles of 16 over head dim 80) ----
        #pragma unroll
        for (int n5 = 0; n5 < 5; ++n5) {
            const size_t vb = (size_t)(n5 * 16 + l15) * SEQ + j + koffB;
            const v16bf vf = mk_frag(*reinterpret_cast<const v8bf*>(&vh[vb]),
                                     *reinterpret_cast<const v8bf*>(&vh[vb + 16]));
            acc[n5] = wmma_bf16(pf, vf, acc[n5]);
        }
    }

    // ---- normalize + store y (bf16, [S, D] with head h at cols h*80..) ----
    #pragma unroll
    for (int r = 0; r < 8; ++r) {
        const float invl = 1.f / lrow[r];
        const int row = m0 + r + (hi << 3);
        #pragma unroll
        for (int n5 = 0; n5 < 5; ++n5) {
            Y[(size_t)row * DIM_ + h * HD + n5 * 16 + l15] = (__bf16)(acc[n5][r] * invl);
        }
    }
}

// ---------------------------------------------------------------------------
// Launcher
// ---------------------------------------------------------------------------
extern "C" void kernel_launch(void* const* d_in, const int* in_sizes, int n_in,
                              void* d_out, int out_size, void* d_ws, size_t ws_size,
                              hipStream_t stream) {
    const float* x      = (const float*)d_in[0];
    const int*   pos    = (const int*)d_in[1];
    // d_in[2] = mask (tril) -- causal mask computed analytically in-kernel
    const float* ln_w   = (const float*)d_in[3];
    const float* ln_b   = (const float*)d_in[4];
    const float* wqkv_w = (const float*)d_in[5];
    const float* wqkv_b = (const float*)d_in[6];
    const float* wo_w   = (const float*)d_in[7];
    const float* wo_b   = (const float*)d_in[8];
    const float* w1_w   = (const float*)d_in[9];
    const float* w1_b   = (const float*)d_in[10];
    const float* w2_w   = (const float*)d_in[11];
    const float* w2_b   = (const float*)d_in[12];
    float* out = (float*)d_out;

    char* ws = (char*)d_ws;
    size_t off = 0;
    auto take = [&](size_t bytes) -> char* {
        char* p = ws + off;
        off += (bytes + 255) & ~(size_t)255;
        return p;
    };
    __bf16* hB      = (__bf16*)take((size_t)SEQ * DIM_ * 2);          // LN(x) bf16
    float*  qkv     = (float*) take((size_t)SEQ * 3 * DIM_ * 4);      // QKV f32
    __bf16* qB      = (__bf16*)take((size_t)NHEAD * SEQ * HDP * 2);   // Q bf16 padded
    __bf16* kB      = (__bf16*)take((size_t)NHEAD * SEQ * HDP * 2);   // K bf16 padded
    __bf16* vT      = (__bf16*)take((size_t)NHEAD * HD * SEQ * 2);    // V^T bf16
    __bf16* yB      = (__bf16*)take((size_t)SEQ * DIM_ * 2);          // attn y bf16
    float*  attnOut = (float*) take((size_t)SEQ * DIM_ * 4);          // WO out f32
    __bf16* gB      = (__bf16*)take((size_t)SEQ * INTER_ * 2);        // gelu(ffn1) bf16
    // total ~183 MB of d_ws

    // 1) LayerNorm -> bf16 h
    ln_kernel<<<dim3(SEQ), dim3(256), 0, stream>>>(x, ln_w, ln_b, hB);
    // 2) QKV = h @ Wqkv^T + b   (f32 out)
    gemm_kernel<0><<<dim3(3 * DIM_ / 128, SEQ / 128), 256, 0, stream>>>(
        hB, wqkv_w, wqkv_b, qkv, nullptr, nullptr, SEQ, 3 * DIM_, DIM_);
    // 3) RoPE + head-major bf16 layouts
    rope_kernel<<<dim3(SEQ, NHEAD), dim3(128), 0, stream>>>(qkv, pos, qB, kB, vT);
    // 4) causal flash attention -> y bf16
    attn_kernel<<<dim3(SEQ / 128, NHEAD), 256, 0, stream>>>(qB, kB, vT, yB);
    // 5) attn_out = y @ Wo^T + b  (f32)
    gemm_kernel<0><<<dim3(DIM_ / 128, SEQ / 128), 256, 0, stream>>>(
        yB, wo_w, wo_b, attnOut, nullptr, nullptr, SEQ, DIM_, DIM_);
    // 6) g = gelu(h @ W1^T + b)  (bf16)
    gemm_kernel<1><<<dim3(INTER_ / 128, SEQ / 128), 256, 0, stream>>>(
        hB, w1_w, w1_b, gB, nullptr, nullptr, SEQ, INTER_, DIM_);
    // 7) out = g @ W2^T + b + attn_out + x   (fused residual epilogue)
    gemm_kernel<2><<<dim3(DIM_ / 128, SEQ / 128), 256, 0, stream>>>(
        gB, w2_w, w2_b, out, attnOut, x, SEQ, DIM_, INTER_);
}
